// SelfAttention2d_21638045237837
// MI455X (gfx1250) — matrix operand order repaired and checked by reading, not compiled, not run on hardware
//
#include <hip/hip_runtime.h>
#include <hip/hip_bf16.h>

// ---------------------------------------------------------------------------
// SelfAttention2d for MI455X (gfx1250, wave32, WMMA + TDM).
// B=4, C=256, CK=32, N=64*64=4096.
//   k = Wk x + bk   [B,32,N]      q = Wq x + bq   [B,32,N]
//   v = Wv x + bv   [B,256,N]
//   affinity[i,j] = sum_ck q[ck,i]*k[ck,j];  softmax over i (axis=1)
//   out[c,j] = sum_i v[c,i]*P[i,j];  result = gamma*out + x
// Streaming (flash) softmax over i, WMMA bf16 16x16x32 everywhere.
// K j-tile is DMA'd to LDS once per block via the Tensor Data Mover.
// ---------------------------------------------------------------------------

#define CH    256
#define CKD   32
#define NPIX  4096
#define NBAT  4

typedef __attribute__((ext_vector_type(16))) __bf16 v16bf;
typedef __attribute__((ext_vector_type(8)))  float  v8f;
typedef __attribute__((ext_vector_type(4)))  unsigned int v4u;
typedef __attribute__((ext_vector_type(8)))  int    v8i;
typedef __attribute__((ext_vector_type(4)))  int    v4i;

union BF16x16 { v16bf v; uint4 q[2]; };

__device__ __forceinline__ v8f v8f_zero() {
    v8f z;
#pragma unroll
    for (int i = 0; i < 8; ++i) z[i] = 0.0f;
    return z;
}

__device__ __forceinline__ v8f wmma_bf16(v16bf a, v16bf b, v8f c) {
    // D = A(16x32 bf16) * B(32x16 bf16) + C(16x16 f32)
    return __builtin_amdgcn_wmma_f32_16x16x32_bf16(
        false, a, false, b, (short)0, c, false, false);
}

// ---------------------------------------------------------------------------
// Kernel 1: fused projections. grid = (N/64, B), block = 320 (10 waves).
// Wave 0 -> k rows 0..31, wave 1 -> q rows 0..31, waves 2..9 -> v rows.
// Outputs: Kt,Qt as [B][N][32] bf16 (row = position, contiguous ck),
//          V as [B][C][N] bf16 (row = channel, contiguous n).
// ---------------------------------------------------------------------------
__global__ __launch_bounds__(320) void proj_kernel(
    const float* __restrict__ x,
    const float* __restrict__ Wk, const float* __restrict__ bk,
    const float* __restrict__ Wq, const float* __restrict__ bq,
    const float* __restrict__ Wv, const float* __restrict__ bv,
    __bf16* __restrict__ Kt, __bf16* __restrict__ Qt, __bf16* __restrict__ V)
{
    const int b    = blockIdx.y;
    const int n0   = blockIdx.x * 64;
    const int tid  = threadIdx.x;
    const int wave = tid >> 5;
    const int lane = tid & 31;
    const int col  = lane & 15;   // column within 16x16 tile
    const int half = lane >> 4;   // lane half selects K sub-chunk

    // transposed X tile: [n][c_local], padded row stride 40 bf16 (80B, 16B-aligned)
    __shared__ __attribute__((aligned(16))) __bf16 xt[64][40];

    const float* Wmat; const float* bias; int row0;
    if      (wave == 0) { Wmat = Wk; bias = bk; row0 = 0; }
    else if (wave == 1) { Wmat = Wq; bias = bq; row0 = 0; }
    else                { Wmat = Wv; bias = bv; row0 = (wave - 2) * 32; }

    v8f acc[2][4];
#pragma unroll
    for (int m = 0; m < 2; ++m)
#pragma unroll
        for (int nn = 0; nn < 4; ++nn) acc[m][nn] = v8f_zero();

    const float* xb = x + (size_t)b * CH * NPIX;

    for (int kc = 0; kc < CH; kc += 32) {
        __syncthreads();
        if (tid < 256) {                       // 8 waves stage, wave-uniform guard
            const int cl = tid & 31;           // c_local 0..31
            const int ng = tid >> 5;           // n group 0..7
            const float* src = xb + (size_t)(kc + cl) * NPIX + n0 + ng * 8;
            float vals[8];
            *(float4*)(vals)     = *(const float4*)(src);
            *(float4*)(vals + 4) = *(const float4*)(src + 4);
            const int nl = ng * 8;
#pragma unroll
            for (int jj = 0; jj < 8; ++jj) xt[nl + jj][cl] = (__bf16)vals[jj];
        }
        __syncthreads();

#pragma unroll
        for (int m = 0; m < 2; ++m) {
            // A fragment from weight rows (f32 -> bf16), contiguous-k layout:
            // lane: row = row0+16m+col, elems k = 16h+0..15
            const int wr = row0 + m * 16 + col;
            const float* wsrc = Wmat + (size_t)wr * CH + kc;
            float4 c0 = *(const float4*)(wsrc + half * 16);
            float4 c1 = *(const float4*)(wsrc + half * 16 + 4);
            float4 c2 = *(const float4*)(wsrc + half * 16 + 8);
            float4 c3 = *(const float4*)(wsrc + half * 16 + 12);
            BF16x16 a;
            a.v[0]=(__bf16)c0.x; a.v[1]=(__bf16)c0.y; a.v[2]=(__bf16)c0.z; a.v[3]=(__bf16)c0.w;
            a.v[4]=(__bf16)c1.x; a.v[5]=(__bf16)c1.y; a.v[6]=(__bf16)c1.z; a.v[7]=(__bf16)c1.w;
            a.v[8]=(__bf16)c2.x; a.v[9]=(__bf16)c2.y; a.v[10]=(__bf16)c2.z; a.v[11]=(__bf16)c2.w;
            a.v[12]=(__bf16)c3.x; a.v[13]=(__bf16)c3.y; a.v[14]=(__bf16)c3.z; a.v[15]=(__bf16)c3.w;
#pragma unroll
            for (int nn = 0; nn < 4; ++nn) {
                // B fragment: lane col n, elems k = e + 16h (contiguous in LDS row)
                const __bf16* bsrc = &xt[nn * 16 + col][half * 16];
                BF16x16 bf;
                bf.q[0] = *(const uint4*)(bsrc);
                bf.q[1] = *(const uint4*)(bsrc + 8);
                acc[m][nn] = wmma_bf16(a.v, bf.v, acc[m][nn]);
            }
        }
    }

    // Epilogue: add bias, store. C/D layout: lane col n, row = 16m + 8h + r.
#pragma unroll
    for (int m = 0; m < 2; ++m) {
#pragma unroll
        for (int nn = 0; nn < 4; ++nn) {
            const int n = n0 + nn * 16 + col;
            if (wave < 2) {
                __attribute__((aligned(16))) __bf16 tmp[8];
#pragma unroll
                for (int r = 0; r < 8; ++r)
                    tmp[r] = (__bf16)(acc[m][nn][r] + bias[m * 16 + half * 8 + r]);
                __bf16* dst = (wave == 0 ? Kt : Qt)
                            + ((size_t)b * NPIX + n) * CKD + m * 16 + half * 8;
                *(uint4*)dst = *(const uint4*)tmp;   // 8 bf16 packed store
            } else {
#pragma unroll
                for (int r = 0; r < 8; ++r) {
                    const int c = row0 + m * 16 + half * 8 + r;
                    V[((size_t)b * CH + c) * NPIX + n] = (__bf16)(acc[m][nn][r] + bias[c]);
                }
            }
        }
    }
}

// ---------------------------------------------------------------------------
// Kernel 2: streaming-softmax attention. grid = (N/32, B), block = 128 (4 waves).
// Each wave owns 64 channels (4 m-subtiles) and redundantly computes the
// 32x32 logit tile S (contraction over ck=32). The j-tile of Kt (32x32 bf16,
// reused by all 128 i-steps) is loaded to LDS once via the Tensor Data Mover.
// ---------------------------------------------------------------------------
__global__ __launch_bounds__(128) void attn_kernel(
    const __bf16* __restrict__ Kt, const __bf16* __restrict__ Qt,
    const __bf16* __restrict__ V,  const float* __restrict__ x,
    const float* __restrict__ gamma, float* __restrict__ out)
{
    const int b    = blockIdx.y;
    const int j0   = blockIdx.x * 32;
    const int tid  = threadIdx.x;
    const int wave = tid >> 5;
    const int lane = tid & 31;
    const int col  = lane & 15;
    const int half = lane >> 4;

    // per-wave scratch to relayout P from C/D layout into B-operand layout
    __shared__ __attribute__((aligned(16))) __bf16 pbuf[4][2][32][16];
    // TDM destination: K rows j0..j0+31, all 32 ck  (row-contiguous, 64B rows)
    __shared__ __attribute__((aligned(16))) __bf16 ktile[32][32];

    const __bf16* Ktb = Kt + (size_t)b * NPIX * CKD;
    const __bf16* Qtb = Qt + (size_t)b * NPIX * CKD;
    const __bf16* Vb  = V  + (size_t)b * CH * NPIX;

    // ---- Tensor Data Mover: DMA the 32x32 bf16 K tile into LDS ----------
    if (wave == 0) {
        const unsigned long long ga =
            (unsigned long long)(uintptr_t)(Ktb + (size_t)j0 * CKD);
        const unsigned int lds_off = (unsigned int)(uintptr_t)(&ktile[0][0]);

        v4u g0;
        g0[0] = 1u;                                  // count=1 (valid), user mode
        g0[1] = lds_off;                             // lds_addr (bytes)
        g0[2] = (unsigned int)(ga & 0xFFFFFFFFu);    // global_addr[31:0]
        g0[3] = (unsigned int)((ga >> 32) & 0x01FFFFFFu) | (2u << 30); // [56:32] | type=2

        v8i g1;
        g1[0] = (1 << 16);        // data_size = 1 (2 bytes), no multicast/pad/iterate
        g1[1] = (32 << 16);       // tensor_dim0[15:0] = 32  (bits 63:48)
        g1[2] = (int)(4096u << 16); // tensor_dim0[31:16]=0 | tensor_dim1[15:0]=4096
        g1[3] = (32 << 16);       // tensor_dim1[31:16]=0 | tile_dim0 = 32
        g1[4] = 32;               // tile_dim1 = 32 | tile_dim2 = 0
        g1[5] = 32;               // tensor_dim0_stride[31:0] = 32 elements
        g1[6] = 0;                // stride0[47:32]=0 | dim1_stride[15:0]=0
        g1[7] = 0;

        v4i z4;
        z4[0] = 0; z4[1] = 0; z4[2] = 0; z4[3] = 0;  // groups 2/3 unused (2-D)
        v8i z8;
#pragma unroll
        for (int i = 0; i < 8; ++i) z8[i] = 0;

        // 6-arg toolchain form: (g0, g1, g2, g3, extra, cpol)
        __builtin_amdgcn_tensor_load_to_lds(g0, g1, z4, z4, z8, 0);
        __builtin_amdgcn_s_wait_tensorcnt((short)0);
    }
    __syncthreads();

    // Hoist the K-side B fragments for this j-tile from LDS.
    BF16x16 bK[2];
#pragma unroll
    for (int js = 0; js < 2; ++js) {
        const __bf16* src = &ktile[js * 16 + col][half * 16];
        bK[js].q[0] = *(const uint4*)(src);
        bK[js].q[1] = *(const uint4*)(src + 8);
    }

    v8f acc[4][2];
#pragma unroll
    for (int ms = 0; ms < 4; ++ms)
#pragma unroll
        for (int js = 0; js < 2; ++js) acc[ms][js] = v8f_zero();

    float mrun[2] = { -__builtin_inff(), -__builtin_inff() };
    float srun[2] = { 0.0f, 0.0f };

    for (int i0 = 0; i0 < NPIX; i0 += 32) {
        // Q-side A fragments: row = i, contiguous ck chunks, k = 16h + e
        BF16x16 aQ[2];
#pragma unroll
        for (int is = 0; is < 2; ++is) {
            const __bf16* src = Qtb + (size_t)(i0 + is * 16 + col) * CKD;
            aQ[is].q[0] = *(const uint4*)(src + half * 16);
            aQ[is].q[1] = *(const uint4*)(src + half * 16 + 8);
        }
        // prefetch next i-step's Q rows into cache (global_prefetch_b8)
        if (i0 + 32 < NPIX)
            __builtin_prefetch(Qtb + (size_t)(i0 + 32 + lane) * CKD, 0, 3);

#pragma unroll
        for (int js = 0; js < 2; ++js) {
            v8f s0 = wmma_bf16(aQ[0].v, bK[js].v, v8f_zero());  // rows i' 0..15
            v8f s1 = wmma_bf16(aQ[1].v, bK[js].v, v8f_zero());  // rows i' 16..31

            // ---- streaming softmax over the i axis (per column j) ----
            float mp = s0[0];
#pragma unroll
            for (int r = 0; r < 8; ++r) { mp = fmaxf(mp, s0[r]); mp = fmaxf(mp, s1[r]); }
            const float cmax = fmaxf(mp, __shfl_xor(mp, 16, 32));
            const float mnew = fmaxf(mrun[js], cmax);
            const float corr = __expf(mrun[js] - mnew);

            float p0[8], p1[8];
            float psum = 0.0f;
#pragma unroll
            for (int r = 0; r < 8; ++r) {
                p0[r] = __expf(s0[r] - mnew);
                p1[r] = __expf(s1[r] - mnew);
                psum += p0[r] + p1[r];
            }
            psum += __shfl_xor(psum, 16, 32);
            srun[js] = srun[js] * corr + psum;
            mrun[js] = mnew;

#pragma unroll
            for (int ms = 0; ms < 4; ++ms)
#pragma unroll
                for (int r = 0; r < 8; ++r) acc[ms][js][r] *= corr;

            // Relayout P (C/D layout -> B layout) through per-wave LDS:
            // src value (tile t, row 8h+r, col) -> dest lane col+16t, elem 8h+r
#pragma unroll
            for (int t = 0; t < 2; ++t) {
                const float* pp = t ? p1 : p0;
#pragma unroll
                for (int r = 0; r < 8; r += 2) {
                    union { __bf16 h[2]; unsigned int u; } pk;
                    pk.h[0] = (__bf16)pp[r];
                    pk.h[1] = (__bf16)pp[r + 1];
                    *(unsigned int*)&pbuf[wave][js][col + 16 * t][half * 8 + r] = pk.u;
                }
            }
        }

        // same-wave LDS is in-order per ISA; make it explicit
        asm volatile("s_wait_dscnt 0" ::: "memory");

        BF16x16 bP[2];
#pragma unroll
        for (int js = 0; js < 2; ++js) {
            const uint4* src = (const uint4*)&pbuf[wave][js][lane][0];
            bP[js].q[0] = src[0];
            bP[js].q[1] = src[1];
        }

        // O[c][j] += V[c][i'] * P[i'][j]
#pragma unroll
        for (int ms = 0; ms < 4; ++ms) {
            const int c = wave * 64 + ms * 16 + col;
            const __bf16* vsrc = Vb + (size_t)c * NPIX + i0;
            BF16x16 aV;
            aV.q[0] = *(const uint4*)(vsrc + half * 16);
            aV.q[1] = *(const uint4*)(vsrc + half * 16 + 8);
            if (i0 + 32 < NPIX)
                __builtin_prefetch(vsrc + 32, 0, 3);  // next 64B segment of this row
#pragma unroll
            for (int js = 0; js < 2; ++js)
                acc[ms][js] = wmma_bf16(aV.v, bP[js].v, acc[ms][js]);
        }
    }

    // Epilogue: out = gamma * (acc / srun) + x
    const float g = gamma[0];
#pragma unroll
    for (int js = 0; js < 2; ++js) {
        const float inv = 1.0f / srun[js];
        const int j = j0 + js * 16 + col;
#pragma unroll
        for (int ms = 0; ms < 4; ++ms) {
#pragma unroll
            for (int r = 0; r < 8; ++r) {
                const int c = wave * 64 + ms * 16 + half * 8 + r;
                const size_t idx = ((size_t)b * CH + c) * NPIX + j;
                out[idx] = g * (acc[ms][js][r] * inv) + x[idx];
            }
        }
    }
}

// ---------------------------------------------------------------------------
extern "C" void kernel_launch(void* const* d_in, const int* in_sizes, int n_in,
                              void* d_out, int out_size, void* d_ws, size_t ws_size,
                              hipStream_t stream)
{
    const float* x     = (const float*)d_in[0];
    const float* Wk    = (const float*)d_in[1];
    const float* bk    = (const float*)d_in[2];
    const float* Wq    = (const float*)d_in[3];
    const float* bq    = (const float*)d_in[4];
    const float* Wv    = (const float*)d_in[5];
    const float* bv    = (const float*)d_in[6];
    const float* gamma = (const float*)d_in[7];
    float* out = (float*)d_out;

    // workspace: Kt (1MB) | Qt (1MB) | V (8MB), all bf16
    __bf16* Kt = (__bf16*)d_ws;
    __bf16* Qt = (__bf16*)((char*)d_ws + (size_t)NBAT * NPIX * CKD * 2);
    __bf16* V  = (__bf16*)((char*)d_ws + (size_t)2 * NBAT * NPIX * CKD * 2);

    proj_kernel<<<dim3(NPIX / 64, NBAT), 320, 0, stream>>>(
        x, Wk, bk, Wq, bq, Wv, bv, Kt, Qt, V);
    attn_kernel<<<dim3(NPIX / 32, NBAT), 128, 0, stream>>>(
        Kt, Qt, V, x, gamma, out);
}
